// Model_46291157516577
// MI455X (gfx1250) — compile-verified, matrix operand order checked by reading, not run on hardware
//
#include <hip/hip_runtime.h>
#include <hip/hip_bf16.h>
#include <math.h>

typedef __attribute__((ext_vector_type(16))) _Float16 v16h;
typedef __attribute__((ext_vector_type(8)))  _Float16 v8h;
typedef __attribute__((ext_vector_type(8)))  float    v8f;

#define T_SEQ 128
#define BATCH 64
#define EMB   1024
#define HID   2048
#define VOC   32000

// ---------------------------------------------------------------------------
// Fragment load for 16x32 f16 WMMA operand from a row-major matrix.
// Per CDNA5 ISA 7.12.2: lanes 0-15 hold rows 0-15 with K={0..7,16..23},
// lanes 16-31 hold rows 0-15 with K={8..15,24..31}.  Since B = W^T with W
// row-major [N x K], a column of B is a contiguous row of W, so the same
// loader serves both A and B operands.
// ---------------------------------------------------------------------------
__device__ __forceinline__ v16h load_frag(const _Float16* __restrict__ src,
                                          int ld, int row0, int k0) {
  const int lane = threadIdx.x & 31;
  const int r  = lane & 15;
  const int hh = lane >> 4;
  const _Float16* p = src + (size_t)(row0 + r) * (size_t)ld + (k0 + hh * 8);
  v8h lo = *(const v8h*)(p);
  v8h hi = *(const v8h*)(p + 16);
  return __builtin_shufflevector(lo, hi, 0, 1, 2, 3, 4, 5, 6, 7,
                                 8, 9, 10, 11, 12, 13, 14, 15);
}

#define WMMA_F16(a, b, c) \
  __builtin_amdgcn_wmma_f32_16x16x32_f16(false, (a), false, (b), (short)0, (c), false, false)

// ---------------------------------------------------------------------------
// fp32 -> f16 conversion (weights, initial h)
// ---------------------------------------------------------------------------
__global__ void f32_to_f16(const float* __restrict__ src,
                           _Float16* __restrict__ dst, size_t n) {
  size_t i = (size_t)blockIdx.x * blockDim.x + threadIdx.x;
  size_t stride = (size_t)gridDim.x * blockDim.x;
  for (; i < n; i += stride) dst[i] = (_Float16)src[i];
}

// ---------------------------------------------------------------------------
// Embedding gather -> f16 activations  xe[tb][e] = embed[x[tb]][e]
// ---------------------------------------------------------------------------
__global__ void embed_gather_f16(const float* __restrict__ embed,
                                 const int* __restrict__ x,
                                 _Float16* __restrict__ out, int TB, int E) {
  size_t n = (size_t)TB * E;
  size_t stride = (size_t)gridDim.x * blockDim.x;
  for (size_t i = (size_t)blockIdx.x * blockDim.x + threadIdx.x; i < n; i += stride) {
    int tb = (int)(i / E);
    int e  = (int)(i - (size_t)tb * E);
    out[i] = (_Float16)embed[(size_t)x[tb] * E + e];
  }
}

// ---------------------------------------------------------------------------
// C[M x N] (fp32) = A[M x K](f16) * W[N x K](f16)^T + bias[N]
// One wave computes a 32x64 block: 2 A-frags x 4 B-frags -> 8 WMMAs per
// K-chunk (0.75 fragment loads per WMMA).
// ---------------------------------------------------------------------------
__global__ void __launch_bounds__(256)
gemm_a16w16_f32(const _Float16* __restrict__ A, const _Float16* __restrict__ W,
                const float* __restrict__ bias, float* __restrict__ C,
                int M, int N, int K) {
  const int gw = (int)((blockIdx.x * (size_t)blockDim.x + threadIdx.x) >> 5);
  const int strips = N >> 6;
  const int mt = gw / strips;
  const int ns = gw - mt * strips;
  if ((mt << 5) >= M) return;
  const int row0 = mt << 5;
  const int col0 = ns << 6;

  v8f acc0[4] = {};
  v8f acc1[4] = {};
  for (int k = 0; k < K; k += 32) {
    v16h a0 = load_frag(A, K, row0, k);
    v16h a1 = load_frag(A, K, row0 + 16, k);
#pragma unroll
    for (int j = 0; j < 4; ++j) {
      v16h b = load_frag(W, K, col0 + 16 * j, k);
      acc0[j] = WMMA_F16(a0, b, acc0[j]);
      acc1[j] = WMMA_F16(a1, b, acc1[j]);
    }
  }

  const int lane = threadIdx.x & 31;
  const int r  = lane & 15;
  const int hh = lane >> 4;
#pragma unroll
  for (int j = 0; j < 4; ++j) {
    const int col = col0 + 16 * j + r;
    const float bv = bias ? bias[col] : 0.0f;
#pragma unroll
    for (int i = 0; i < 8; ++i) {
      C[(size_t)(row0 + i + 8 * hh) * (size_t)N + col]      = acc0[j][i] + bv;
      C[(size_t)(row0 + 16 + i + 8 * hh) * (size_t)N + col] = acc1[j][i] + bv;
    }
  }
}

// ---------------------------------------------------------------------------
// Fused LSTM step:  gates = gx_t + h_prev * Wh^T + bh ; cell update.
// Grid: H/16 blocks.  Block: 1024 threads = 32 waves.
// K (== H) is split in two slices across the 32 waves to double wave-level
// parallelism in the serial (per-timestep) region:
//   wave w: kslice = w>>4, gate g = (w&15)>>2, batch tile mt = w&3.
// Each wave accumulates one 16x16 tile over K/2; partials land in LDS, the
// pointwise phase sums the slices and applies the sigmoid/tanh cell update.
// ---------------------------------------------------------------------------
__global__ void __launch_bounds__(1024)
lstm_step(const _Float16* __restrict__ hprev,   // [64 x H] f16
          const _Float16* __restrict__ Wh,      // [4H x H] f16 row-major
          const float* __restrict__ gx_t,       // [64 x 4H] fp32 (incl. bx)
          const float* __restrict__ bh,         // [4H]
          float* __restrict__ c,                // [64 x H] in/out fp32
          _Float16* __restrict__ y_t,           // [64 x H] f16 (h_t)
          float* __restrict__ h32,              // [64 x H] fp32 (h_t)
          int H) {
  const int w   = threadIdx.x >> 5;      // 0..31
  const int ks  = w >> 4;                // K slice: 0 or 1
  const int wl  = w & 15;
  const int g   = wl >> 2;               // gate: 0=i 1=f 2=o 3=n
  const int mt  = wl & 3;                // batch tile
  const int jb  = blockIdx.x << 4;       // hidden column block
  const int row0 = mt << 4;
  const int col0 = g * H + jb;
  const int Kh   = H >> 1;
  const int kbeg = ks * Kh;

  v8f acc = {};
  for (int k = kbeg; k < kbeg + Kh; k += 32) {
    v16h a = load_frag(hprev, H, row0, k);
    v16h b = load_frag(Wh, H, col0, k);
    acc = WMMA_F16(a, b, acc);
  }

  __shared__ float gl[2][4][64][16];     // [kslice][gate][batch][col] = 32 KB
  const int lane = threadIdx.x & 31;
  const int r  = lane & 15;
  const int hh = lane >> 4;
#pragma unroll
  for (int i = 0; i < 8; ++i) {
    gl[ks][g][row0 + i + 8 * hh][r] = acc[i];
  }
  __syncthreads();

  const int H4 = H << 2;
  for (int e = threadIdx.x; e < BATCH * 16; e += 1024) {
    const int b   = e >> 4;
    const int col = e & 15;
    const int jc  = jb + col;
    const size_t gxb = (size_t)b * H4 + jc;
    const float xi = gl[0][0][b][col] + gl[1][0][b][col] + gx_t[gxb        ] + bh[jc        ];
    const float xf = gl[0][1][b][col] + gl[1][1][b][col] + gx_t[gxb +     H] + bh[jc +     H];
    const float xo = gl[0][2][b][col] + gl[1][2][b][col] + gx_t[gxb + 2 * H] + bh[jc + 2 * H];
    const float xn = gl[0][3][b][col] + gl[1][3][b][col] + gx_t[gxb + 3 * H] + bh[jc + 3 * H];
    const size_t idx = (size_t)b * H + jc;
    const float si = 1.0f / (1.0f + __expf(-xi));
    const float sf = 1.0f / (1.0f + __expf(-xf));
    const float so = 1.0f / (1.0f + __expf(-xo));
    const float cn = sf * c[idx] + si * tanhf(xn);
    const float hn = so * tanhf(cn);
    c[idx]   = cn;
    h32[idx] = hn;
    y_t[idx] = (_Float16)hn;
  }
}

// ---------------------------------------------------------------------------
// Host-side orchestration
// ---------------------------------------------------------------------------
static inline void launch_conv(const float* s, _Float16* d, size_t n, hipStream_t st) {
  int blocks = (int)((n + 255) / 256);
  if (blocks > 16384) blocks = 16384;
  f32_to_f16<<<blocks, 256, 0, st>>>(s, d, n);
}

static inline void launch_gemm(const _Float16* A, const _Float16* W, const float* bias,
                               float* C, int M, int N, int K, hipStream_t st) {
  size_t waves = (size_t)(M / 32) * (size_t)(N / 64);
  int blocks = (int)((waves + 7) / 8);
  gemm_a16w16_f32<<<blocks, 256, 0, st>>>(A, W, bias, C, M, N, K);
}

extern "C" void kernel_launch(void* const* d_in, const int* in_sizes, int n_in,
                              void* d_out, int out_size, void* d_ws, size_t ws_size,
                              hipStream_t stream) {
  (void)in_sizes; (void)n_in; (void)out_size; (void)ws_size;

  const int*   x      = (const int*)  d_in[0];
  const float* h0_0   = (const float*)d_in[1];
  const float* c0_0   = (const float*)d_in[2];
  const float* h1_0   = (const float*)d_in[3];
  const float* c1_0   = (const float*)d_in[4];
  const float* h2_0   = (const float*)d_in[5];
  const float* c2_0   = (const float*)d_in[6];
  const float* embedW = (const float*)d_in[7];
  const float* Wx0 = (const float*)d_in[8];
  const float* Wh0 = (const float*)d_in[9];
  const float* bx0 = (const float*)d_in[10];
  const float* bh0 = (const float*)d_in[11];
  const float* Wx1 = (const float*)d_in[12];
  const float* Wh1 = (const float*)d_in[13];
  const float* bx1 = (const float*)d_in[14];
  const float* bh1 = (const float*)d_in[15];
  const float* Wx2 = (const float*)d_in[16];
  const float* Wh2 = (const float*)d_in[17];
  const float* bx2 = (const float*)d_in[18];
  const float* bh2 = (const float*)d_in[19];
  const float* fc_b = (const float*)d_in[20];
  float* out = (float*)d_out;

  const int TB = T_SEQ * BATCH;                       // 8192

  // ---- workspace carve (256B aligned bump allocator) ----
  char* base = (char*)d_ws;
  size_t off = 0;
  auto alloc = [&](size_t bytes) -> void* {
    void* p = base + off;
    off = (off + bytes + 255) & ~(size_t)255;
    return p;
  };

  _Float16* eW16   = (_Float16*)alloc((size_t)VOC * EMB * 2);
  _Float16* Wx0_16 = (_Float16*)alloc((size_t)4 * HID * EMB * 2);
  _Float16* Wh0_16 = (_Float16*)alloc((size_t)4 * HID * HID * 2);
  _Float16* Wx1_16 = (_Float16*)alloc((size_t)4 * HID * HID * 2);
  _Float16* Wh1_16 = (_Float16*)alloc((size_t)4 * HID * HID * 2);
  _Float16* Wx2_16 = (_Float16*)alloc((size_t)4 * EMB * HID * 2);
  _Float16* Wh2_16 = (_Float16*)alloc((size_t)4 * EMB * EMB * 2);
  _Float16* xe16   = (_Float16*)alloc((size_t)TB * EMB * 2);
  _Float16* y0     = (_Float16*)alloc((size_t)TB * HID * 2);
  _Float16* y1     = (_Float16*)alloc((size_t)TB * HID * 2);
  _Float16* y2     = (_Float16*)alloc((size_t)TB * EMB * 2);
  float*    gx     = (float*)   alloc((size_t)TB * 4 * HID * 4);  // reused per layer
  _Float16* h016   = (_Float16*)alloc((size_t)BATCH * HID * 2);   // reused per layer
  float*    cbuf0  = (float*)   alloc((size_t)BATCH * HID * 4);
  float*    cbuf1  = (float*)   alloc((size_t)BATCH * HID * 4);
  float*    cbuf2  = (float*)   alloc((size_t)BATCH * EMB * 4);
  float*    hf0    = (float*)   alloc((size_t)BATCH * HID * 4);
  float*    hf1    = (float*)   alloc((size_t)BATCH * HID * 4);
  float*    hf2    = (float*)   alloc((size_t)BATCH * EMB * 4);

  // ---- weight / activation precision conversion ----
  launch_conv(embedW, eW16,   (size_t)VOC * EMB,     stream);
  launch_conv(Wx0,    Wx0_16, (size_t)4 * HID * EMB, stream);
  launch_conv(Wh0,    Wh0_16, (size_t)4 * HID * HID, stream);
  launch_conv(Wx1,    Wx1_16, (size_t)4 * HID * HID, stream);
  launch_conv(Wh1,    Wh1_16, (size_t)4 * HID * HID, stream);
  launch_conv(Wx2,    Wx2_16, (size_t)4 * EMB * HID, stream);
  launch_conv(Wh2,    Wh2_16, (size_t)4 * EMB * EMB, stream);

  {
    int blocks = (int)(((size_t)TB * EMB + 255) / 256);
    if (blocks > 16384) blocks = 16384;
    embed_gather_f16<<<blocks, 256, 0, stream>>>(embedW, x, xe16, TB, EMB);
  }

  // ================= layer 0 (in=EMB, hidden=HID) =================
  launch_conv(h0_0, h016, (size_t)BATCH * HID, stream);
  hipMemcpyAsync(cbuf0, c0_0, (size_t)BATCH * HID * 4, hipMemcpyDeviceToDevice, stream);
  launch_gemm(xe16, Wx0_16, bx0, gx, TB, 4 * HID, EMB, stream);
  for (int t = 0; t < T_SEQ; ++t) {
    const _Float16* hp = (t == 0) ? h016 : (y0 + (size_t)(t - 1) * BATCH * HID);
    lstm_step<<<HID / 16, 1024, 0, stream>>>(
        hp, Wh0_16, gx + (size_t)t * BATCH * 4 * HID, bh0,
        cbuf0, y0 + (size_t)t * BATCH * HID, hf0, HID);
  }

  // ================= layer 1 (in=HID, hidden=HID) =================
  launch_conv(h1_0, h016, (size_t)BATCH * HID, stream);
  hipMemcpyAsync(cbuf1, c1_0, (size_t)BATCH * HID * 4, hipMemcpyDeviceToDevice, stream);
  launch_gemm(y0, Wx1_16, bx1, gx, TB, 4 * HID, HID, stream);
  for (int t = 0; t < T_SEQ; ++t) {
    const _Float16* hp = (t == 0) ? h016 : (y1 + (size_t)(t - 1) * BATCH * HID);
    lstm_step<<<HID / 16, 1024, 0, stream>>>(
        hp, Wh1_16, gx + (size_t)t * BATCH * 4 * HID, bh1,
        cbuf1, y1 + (size_t)t * BATCH * HID, hf1, HID);
  }

  // ================= layer 2 (in=HID, hidden=EMB) =================
  launch_conv(h2_0, h016, (size_t)BATCH * EMB, stream);
  hipMemcpyAsync(cbuf2, c2_0, (size_t)BATCH * EMB * 4, hipMemcpyDeviceToDevice, stream);
  launch_gemm(y1, Wx2_16, bx2, gx, TB, 4 * EMB, HID, stream);
  for (int t = 0; t < T_SEQ; ++t) {
    const _Float16* hp = (t == 0) ? h016 : (y2 + (size_t)(t - 1) * BATCH * EMB);
    lstm_step<<<EMB / 16, 1024, 0, stream>>>(
        hp, Wh2_16, gx + (size_t)t * BATCH * 4 * EMB, bh2,
        cbuf2, y2 + (size_t)t * BATCH * EMB, hf2, EMB);
  }

  // ================= final tied FC: scores = y2 * embed_W^T + fc_b ==========
  launch_gemm(y2, eW16, fc_b, out, TB, VOC, EMB, stream);

  // ================= final states into d_out tail ===========================
  size_t so = (size_t)TB * VOC;
  hipMemcpyAsync(out + so, hf0,   (size_t)BATCH * HID * 4, hipMemcpyDeviceToDevice, stream); so += (size_t)BATCH * HID;
  hipMemcpyAsync(out + so, cbuf0, (size_t)BATCH * HID * 4, hipMemcpyDeviceToDevice, stream); so += (size_t)BATCH * HID;
  hipMemcpyAsync(out + so, hf1,   (size_t)BATCH * HID * 4, hipMemcpyDeviceToDevice, stream); so += (size_t)BATCH * HID;
  hipMemcpyAsync(out + so, cbuf1, (size_t)BATCH * HID * 4, hipMemcpyDeviceToDevice, stream); so += (size_t)BATCH * HID;
  hipMemcpyAsync(out + so, hf2,   (size_t)BATCH * EMB * 4, hipMemcpyDeviceToDevice, stream); so += (size_t)BATCH * EMB;
  hipMemcpyAsync(out + so, cbuf2, (size_t)BATCH * EMB * 4, hipMemcpyDeviceToDevice, stream);
}
